// InvariantPointAttention_23476291240027
// MI455X (gfx1250) — matrix operand order repaired and checked by reading, not compiled, not run on hardware
//
#include <hip/hip_runtime.h>
#include <hip/hip_bf16.h>
#include <math.h>

// ---------------------------------------------------------------------------
// Invariant Point Attention for MI455X (gfx1250, wave32, WMMA bf16).
// Memory-bound on inputs_2d (2 x 134 MB fp32 streams -> ~11.5us HBM floor at
// 23.3 TB/s). All dense math on v_wmma_f32_16x16x32_bf16; the two inputs_2d
// streaming kernels stage tiles in LDS via GLOBAL_LOAD_ASYNC_TO_LDS_B128
// (ASYNCcnt) with double buffering to overlap DMA with WMMA.
// ---------------------------------------------------------------------------

typedef __bf16 bf16;
typedef __attribute__((ext_vector_type(16))) __bf16 v16bf;
typedef __attribute__((ext_vector_type(8)))  float  v8f;
typedef int v4i __attribute__((vector_size(16)));   // matches builtin param type

#define WMMA_BF16(A, B, C) \
  __builtin_amdgcn_wmma_f32_16x16x32_bf16(false, (A), false, (B), (short)0, (C), false, false)

#define AS1 __attribute__((address_space(1)))
#define AS3 __attribute__((address_space(3)))

#if defined(__has_builtin)
#if __has_builtin(__builtin_amdgcn_global_load_async_to_lds_b128)
#define HAS_ASYNC_LDS 1
#else
#define HAS_ASYNC_LDS 0
#endif
#else
#define HAS_ASYNC_LDS 0
#endif

// 16-byte global -> LDS copy; async (ASYNCcnt) when the toolchain has the
// gfx1250 builtin, synchronous float4 otherwise.
__device__ __forceinline__ void lds_copy16(const float* g, float* l) {
#if HAS_ASYNC_LDS
  __builtin_amdgcn_global_load_async_to_lds_b128((AS1 v4i*)g, (AS3 v4i*)l, 0, 0);
#else
  *(float4*)l = *(const float4*)g;
#endif
}

// Wait for outstanding async LDS traffic, then block barrier.
__device__ __forceinline__ void lds_copy_join() {
#if HAS_ASYNC_LDS
#if __has_builtin(__builtin_amdgcn_s_wait_asynccnt)
  __builtin_amdgcn_s_wait_asynccnt(0);
#else
  asm volatile("s_wait_asynccnt 0x0" ::: "memory");
#endif
#endif
  __syncthreads();
}

// Problem constants
#define NRES   512
#define NHEAD  12
#define C1     384
#define C2     128
#define NPROJ  1152   // 192 (q) + 384 (kv) + 144 (qp) + 432 (kvp)
#define NFIN   2112
#define NOUT   384

static constexpr float SCALAR_W = 0.14433756729740643f;   // sqrt(1/48)
static constexpr float POINT_W0 = 0.13608276348795434f;   // sqrt(1/54)
static constexpr float A2D_W    = 0.5773502691896258f;    // sqrt(1/3)
static constexpr float FP16HUGE = 1.0e9f;

// ---------------------------------------------------------------------------
// Fragment loaders (wave32; contiguous k order shared by A and B, not the ISA table order)
// A (16Mx32K bf16): lane m=lane&15, half=lane>>4; vgpr v holds
//   K = half*16 + 2v (+1)
// B (32Kx16N bf16): lane n=lane&15, half=lane>>4; vgpr v holds
//   K = half*16 + 2v (+1)
// C/D (16x16 f32): reg r -> m = r + half*8, n = lane&15
// ---------------------------------------------------------------------------
__device__ __forceinline__ v16bf load_a_rm_bf(const bf16* src, int lda) {
  int lane = threadIdx.x & 31, m = lane & 15, half = lane >> 4;
  v16bf a;
#pragma unroll
  for (int v = 0; v < 8; ++v) {
    int kb = half * 16 + 2 * v;
    a[2 * v]     = src[(size_t)m * lda + kb];
    a[2 * v + 1] = src[(size_t)m * lda + kb + 1];
  }
  return a;
}

__device__ __forceinline__ v16bf load_a_rm_f32(const float* src, int lda) {
  int lane = threadIdx.x & 31, m = lane & 15, half = lane >> 4;
  v16bf a;
#pragma unroll
  for (int v = 0; v < 8; ++v) {
    int kb = half * 16 + 2 * v;
    a[2 * v]     = (bf16)src[(size_t)m * lda + kb];
    a[2 * v + 1] = (bf16)src[(size_t)m * lda + kb + 1];
  }
  return a;
}

__device__ __forceinline__ v16bf load_b_rm_bf(const bf16* src, int ldb) {
  int lane = threadIdx.x & 31, n = lane & 15, half = lane >> 4;
  v16bf b;
#pragma unroll
  for (int v = 0; v < 8; ++v) {
    int k = half * 16 + 2 * v;
    b[2 * v]     = src[(size_t)k * ldb + n];
    b[2 * v + 1] = src[(size_t)(k + 1) * ldb + n];
  }
  return b;
}

__device__ __forceinline__ v16bf load_b_rm_f32(const float* src, int ldb) {
  int lane = threadIdx.x & 31, n = lane & 15, half = lane >> 4;
  v16bf b;
#pragma unroll
  for (int v = 0; v < 8; ++v) {
    int k = half * 16 + 2 * v;
    b[2 * v]     = (bf16)src[(size_t)k * ldb + n];
    b[2 * v + 1] = (bf16)src[(size_t)(k + 1) * ldb + n];
  }
  return b;
}

// B from an N-major (N x K row-major) source: B[k][n] = src[n*ld + k]
__device__ __forceinline__ v16bf load_b_cm_bf(const bf16* src, int ld) {
  int lane = threadIdx.x & 31, n = lane & 15, half = lane >> 4;
  v16bf b;
#pragma unroll
  for (int v = 0; v < 8; ++v) {
    int k = half * 16 + 2 * v;
    b[2 * v]     = src[(size_t)n * ld + k];
    b[2 * v + 1] = src[(size_t)n * ld + k + 1];
  }
  return b;
}

// ---------------------------------------------------------------------------
// Prep kernels
// ---------------------------------------------------------------------------
__global__ void k_f2bf(const float* __restrict__ src, bf16* __restrict__ dst, int n) {
  int i = blockIdx.x * blockDim.x + threadIdx.x;
  if (i < n) dst[i] = (bf16)src[i];
}

__global__ void k_build_wall(const float* __restrict__ wq, const float* __restrict__ wkv,
                             const float* __restrict__ wqp, const float* __restrict__ wkvp,
                             bf16* __restrict__ W) {
  int idx = blockIdx.x * blockDim.x + threadIdx.x;
  if (idx >= C1 * NPROJ) return;
  int k = idx / NPROJ, c = idx % NPROJ;
  float v;
  if (c < 192)       v = wq[k * 192 + c];
  else if (c < 576)  v = wkv[k * 384 + (c - 192)];
  else if (c < 720)  v = wqp[k * 144 + (c - 576)];
  else               v = wkvp[k * 432 + (c - 720)];
  W[idx] = (bf16)v;
}

__global__ void k_build_w2d(const float* __restrict__ w2d, bf16* __restrict__ W) {
  int idx = blockIdx.x * blockDim.x + threadIdx.x;
  if (idx >= C2 * 16) return;
  int k = idx / 16, c = idx % 16;
  W[idx] = (bf16)((c < NHEAD) ? w2d[k * NHEAD + c] : 0.0f);
}

__global__ void k_zero_attn_pad(bf16* __restrict__ attn_bf) {
  int idx = blockIdx.x * blockDim.x + threadIdx.x;  // 512 * 4 * 512
  if (idx >= NRES * 4 * NRES) return;
  int i = idx / (4 * NRES);
  int rem = idx % (4 * NRES);
  int hh = NHEAD + rem / NRES;
  int k = rem % NRES;
  attn_bf[((size_t)i * 16 + hh) * NRES + k] = (bf16)0.0f;
}

// ---------------------------------------------------------------------------
// Fused projection GEMM: Y(512 x 1152) = X(512x384) @ Wall(384x1152) + bias
// ---------------------------------------------------------------------------
__global__ void k_proj(const bf16* __restrict__ X, const bf16* __restrict__ W,
                       const float* __restrict__ bq, const float* __restrict__ bkv,
                       const float* __restrict__ bqp, const float* __restrict__ bkvp,
                       float* __restrict__ Y) {
  int wid = (blockIdx.x * blockDim.x + threadIdx.x) >> 5;  // 2304 waves
  int mt = wid / 72, nt = wid % 72;
  if (mt >= 32) return;
  int lane = threadIdx.x & 31, half = lane >> 4;
  v8f acc = {};
#pragma unroll 1
  for (int kc = 0; kc < 12; ++kc) {
    v16bf a = load_a_rm_bf(X + (size_t)(mt * 16) * C1 + kc * 32, C1);
    v16bf b = load_b_rm_bf(W + (size_t)(kc * 32) * NPROJ + nt * 16, NPROJ);
    acc = WMMA_BF16(a, b, acc);
  }
  int n = nt * 16 + (lane & 15);
  float bias = (n < 192) ? bq[n] : (n < 576) ? bkv[n - 192]
             : (n < 720) ? bqp[n - 576] : bkvp[n - 720];
#pragma unroll
  for (int r = 0; r < 8; ++r) {
    int m = mt * 16 + r + half * 8;
    Y[(size_t)m * NPROJ + n] = acc[r] + bias;
  }
}

// ---------------------------------------------------------------------------
// to_global + build packed attention operands.
// QA[h][n][32]: [0:16)=SCALAR_W*q_scalar, [16:28)=pw*q_point_global, [28:32)=0
// KB[h][n][32]: [0:16)=k_scalar,          [16:28)=k_point_global,    [28:32)=0
// Gq/Gk[h][n] = sum over 12 point dims of global-point^2
// Vs[h][k][16]=v_scalar ; Vp[h][k][32] = v_point_global (24 used, pad 0)
// ---------------------------------------------------------------------------
__global__ void k_post_proj(const float* __restrict__ Y, const float* __restrict__ rot,
                            const float* __restrict__ trans, const float* __restrict__ tpw,
                            bf16* __restrict__ QA, bf16* __restrict__ KB,
                            float* __restrict__ Gq, float* __restrict__ Gk,
                            bf16* __restrict__ Vs, bf16* __restrict__ Vp) {
  int idx = blockIdx.x * blockDim.x + threadIdx.x;
  if (idx >= NHEAD * NRES) return;
  int h = idx >> 9, n = idx & 511;
  float R[3][3], T[3];
#pragma unroll
  for (int d = 0; d < 3; ++d) {
    T[d] = trans[n * 3 + d];
#pragma unroll
    for (int j = 0; j < 3; ++j) R[d][j] = rot[n * 9 + d * 3 + j];
  }
  float pw = POINT_W0 * log1pf(expf(tpw[h]));
  const float* y = Y + (size_t)n * NPROJ;

  bf16* qa = QA + ((size_t)h * NRES + n) * 32;
  bf16* kb = KB + ((size_t)h * NRES + n) * 32;
#pragma unroll
  for (int c = 0; c < 16; ++c) {
    qa[c] = (bf16)(SCALAR_W * y[h * 16 + c]);
    kb[c] = (bf16)(y[192 + h * 32 + c]);
  }
  float gq = 0.0f, gk = 0.0f;
#pragma unroll
  for (int p = 0; p < 4; ++p) {
    float lq[3], lk[3];
#pragma unroll
    for (int j = 0; j < 3; ++j) {
      lq[j] = y[576 + j * 48 + h * 4 + p];
      lk[j] = y[720 + j * 144 + h * 12 + p];
    }
#pragma unroll
    for (int d = 0; d < 3; ++d) {
      float gqd = R[d][0] * lq[0] + R[d][1] * lq[1] + R[d][2] * lq[2] + T[d];
      float gkd = R[d][0] * lk[0] + R[d][1] * lk[1] + R[d][2] * lk[2] + T[d];
      qa[16 + p * 3 + d] = (bf16)(pw * gqd);
      kb[16 + p * 3 + d] = (bf16)gkd;
      gq += gqd * gqd;
      gk += gkd * gkd;
    }
  }
#pragma unroll
  for (int c = 28; c < 32; ++c) { qa[c] = (bf16)0.0f; kb[c] = (bf16)0.0f; }
  Gq[h * NRES + n] = gq;
  Gk[h * NRES + n] = gk;

  bf16* vs = Vs + ((size_t)h * NRES + n) * 16;
#pragma unroll
  for (int c = 0; c < 16; ++c) vs[c] = (bf16)(y[192 + h * 32 + 16 + c]);

  bf16* vp = Vp + ((size_t)h * NRES + n) * 32;
#pragma unroll
  for (int p = 0; p < 8; ++p) {
    float l[3];
#pragma unroll
    for (int j = 0; j < 3; ++j) l[j] = y[720 + j * 144 + h * 12 + 4 + p];
#pragma unroll
    for (int d = 0; d < 3; ++d)
      vp[p * 3 + d] = (bf16)(R[d][0] * l[0] + R[d][1] * l[1] + R[d][2] * l[2] + T[d]);
  }
#pragma unroll
  for (int c = 24; c < 32; ++c) vp[c] = (bf16)0.0f;
}

// ---------------------------------------------------------------------------
// a2d[h][r] = A2D_W * (inputs_2d[r,:] @ w2d[:,h] + b2d[h]),  r = i*512+j
// First streaming pass over inputs_2d (134 MB). Each block async-stages a
// 128x128 f32 tile (64 KB) into LDS, then 8 waves run 4 WMMAs each from LDS.
// ---------------------------------------------------------------------------
__global__ void k_a2d(const float* __restrict__ in2d, const bf16* __restrict__ W2d,
                      const float* __restrict__ b2d, float* __restrict__ a2d) {
  __shared__ __align__(16) float tile[128 * C2];  // 64 KB
  int wave = threadIdx.x >> 5, lane = threadIdx.x & 31, half = lane >> 4;
  size_t rblk = (size_t)blockIdx.x * 128;  // 2048 blocks cover 262144 rows
  const float* gsrc = in2d + rblk * C2;

  // Stage 128x128 f32 -> LDS with coalesced 16B async copies (16 per lane).
#pragma unroll
  for (int t = threadIdx.x; t < (128 * C2) / 4; t += 256)
    lds_copy16(gsrc + (size_t)t * 4, &tile[t * 4]);
  lds_copy_join();

  const float* Ab = &tile[wave * 16 * C2];
  v8f acc = {};
#pragma unroll
  for (int kc = 0; kc < 4; ++kc) {
    v16bf a = load_a_rm_f32(Ab + kc * 32, C2);
    v16bf b = load_b_rm_bf(W2d + (size_t)(kc * 32) * 16, 16);
    acc = WMMA_BF16(a, b, acc);
  }
  int h = lane & 15;
  if (h < NHEAD) {
    float bb = b2d[h];
    size_t r0 = rblk + wave * 16;
#pragma unroll
    for (int r = 0; r < 8; ++r) {
      size_t row = r0 + r + half * 8;
      a2d[(size_t)h * NRES * NRES + row] = A2D_W * (acc[r] + bb);
    }
  }
}

// ---------------------------------------------------------------------------
// Attention core: one block per (head, 16-query tile).
// logits = WMMA(QA,KB^T) - 0.5*pw*(Gq+Gk) + a2d - HUGE*(1-mq*mk)
// then softmax over 512 keys, then fused attn@Vs and attn@Vp WMMAs.
// ---------------------------------------------------------------------------
__global__ void k_attn(const bf16* __restrict__ QA, const bf16* __restrict__ KB,
                       const float* __restrict__ Gq, const float* __restrict__ Gk,
                       const float* __restrict__ a2d, const float* __restrict__ mask,
                       const float* __restrict__ tpw, const bf16* __restrict__ Vs,
                       const bf16* __restrict__ Vp, bf16* __restrict__ attn_bf,
                       bf16* __restrict__ FA, float* __restrict__ RPG) {
  __shared__ float L[16][NRES];
  __shared__ float redM[16][16];
  __shared__ float redS[16][16];

  int h = blockIdx.x >> 5;
  int q0 = (blockIdx.x & 31) * 16;
  int wave = threadIdx.x >> 5, lane = threadIdx.x & 31;
  int half = lane >> 4;
  float pw = POINT_W0 * log1pf(expf(tpw[h]));

  // --- Phase 1: logits (8 waves x 4 key-tiles each) ---
  v16bf afrag = load_a_rm_bf(QA + ((size_t)h * NRES + q0) * 32, 32);
#pragma unroll 1
  for (int kt = wave; kt < 32; kt += 8) {
    int k0 = kt * 16;
    v16bf b = load_b_cm_bf(KB + ((size_t)h * NRES + k0) * 32, 32);
    v8f s = {};
    s = WMMA_BF16(afrag, b, s);
    int kcol = k0 + (lane & 15);
    float gk = Gk[h * NRES + kcol];
    float mk = mask[kcol];
#pragma unroll
    for (int r = 0; r < 8; ++r) {
      int m = r + half * 8;
      int q = q0 + m;
      float val = s[r] - 0.5f * pw * (Gq[h * NRES + q] + gk)
                + a2d[(size_t)h * NRES * NRES + (size_t)q * NRES + kcol]
                - FP16HUGE * (1.0f - mask[q] * mk);
      L[m][kcol] = val;
    }
  }
  __syncthreads();

  // --- Phase 2: row softmax (16 rows x 16 segments of 32 keys) ---
  int row = threadIdx.x >> 4, seg = threadIdx.x & 15;
  float mx = -3.0e38f;
  for (int k = seg * 32; k < seg * 32 + 32; ++k) mx = fmaxf(mx, L[row][k]);
  redM[row][seg] = mx;
  __syncthreads();
  if (seg == 0) {
    float m2 = redM[row][0];
    for (int t = 1; t < 16; ++t) m2 = fmaxf(m2, redM[row][t]);
    redM[row][0] = m2;
  }
  __syncthreads();
  mx = redM[row][0];
  float sum = 0.0f;
  for (int k = seg * 32; k < seg * 32 + 32; ++k) {
    float e = __expf(L[row][k] - mx);
    L[row][k] = e;
    sum += e;
  }
  redS[row][seg] = sum;
  __syncthreads();
  if (seg == 0) {
    float s2 = 0.0f;
    for (int t = 0; t < 16; ++t) s2 += redS[row][t];
    redS[row][0] = s2;
  }
  __syncthreads();
  float inv = 1.0f / redS[row][0];
  for (int k = seg * 32; k < seg * 32 + 32; ++k) {
    float a = L[row][k] * inv;
    L[row][k] = a;
    attn_bf[((size_t)(q0 + row) * 16 + h) * NRES + k] = (bf16)a;
  }
  __syncthreads();

  // --- Phase 3: fused attn @ V (wave 0: scalar V; waves 1,2: point V) ---
  if (wave == 0) {
    v8f acc = {};
#pragma unroll 1
    for (int kc = 0; kc < 16; ++kc) {
      v16bf a = load_a_rm_f32(&L[0][0] + kc * 32, NRES);
      v16bf b = load_b_rm_bf(Vs + ((size_t)h * NRES + kc * 32) * 16, 16);
      acc = WMMA_BF16(a, b, acc);
    }
#pragma unroll
    for (int r = 0; r < 8; ++r) {
      int i = q0 + r + half * 8;
      FA[(size_t)i * NFIN + h * 16 + (lane & 15)] = (bf16)acc[r];
    }
  } else if (wave == 1 || wave == 2) {
    int nt = wave - 1;
    v8f acc = {};
#pragma unroll 1
    for (int kc = 0; kc < 16; ++kc) {
      v16bf a = load_a_rm_f32(&L[0][0] + kc * 32, NRES);
      v16bf b = load_b_rm_bf(Vp + ((size_t)h * NRES + kc * 32) * 32 + nt * 16, 32);
      acc = WMMA_BF16(a, b, acc);
    }
    int pd = nt * 16 + (lane & 15);
#pragma unroll
    for (int r = 0; r < 8; ++r) {
      int i = q0 + r + half * 8;
      if (pd < 24) RPG[((size_t)i * NHEAD + h) * 24 + pd] = acc[r];
    }
  }
}

// ---------------------------------------------------------------------------
// rp_local = rot^T (result_point_global - trans); point_norm = |rp_local|
// ---------------------------------------------------------------------------
__global__ void k_rpl(const float* __restrict__ RPG, const float* __restrict__ rot,
                      const float* __restrict__ trans, bf16* __restrict__ FA) {
  int idx = blockIdx.x * blockDim.x + threadIdx.x;  // 512*96
  if (idx >= NRES * 96) return;
  int i = idx / 96, hp = idx % 96;
  int h = hp / 8, p = hp % 8;
  const float* g = RPG + ((size_t)i * NHEAD + h) * 24 + p * 3;
  float x = g[0] - trans[i * 3 + 0];
  float y = g[1] - trans[i * 3 + 1];
  float z = g[2] - trans[i * 3 + 2];
  const float* R = rot + i * 9;
  float l0 = R[0] * x + R[3] * y + R[6] * z;
  float l1 = R[1] * x + R[4] * y + R[7] * z;
  float l2 = R[2] * x + R[5] * y + R[8] * z;
  float nrm = sqrtf(1e-8f + l0 * l0 + l1 * l1 + l2 * l2);
  size_t base = (size_t)i * NFIN;
  FA[base + 192 + hp] = (bf16)l0;
  FA[base + 288 + hp] = (bf16)l1;
  FA[base + 384 + hp] = (bf16)l2;
  FA[base + 480 + hp] = (bf16)nrm;
}

// ---------------------------------------------------------------------------
// attn_over_2d: per query row i, (16heads x 512) @ inputs_2d[i](512x128).
// Second streaming pass over inputs_2d. One block per i; 8 waves = 8 N-tiles.
// Double-buffered async LDS staging of 32x128 f32 K-chunks (2 x 16 KB):
// chunk k+1 DMAs while WMMA consumes chunk k.
// ---------------------------------------------------------------------------
__global__ void k_over2d(const bf16* __restrict__ attn_bf, const float* __restrict__ in2d,
                         bf16* __restrict__ FA) {
  __shared__ __align__(16) float buf[2][32 * C2];  // 2 x 16 KB
  int i = blockIdx.x;
  int wave = threadIdx.x >> 5, lane = threadIdx.x & 31, half = lane >> 4;
  int c0 = wave * 16;
  const bf16* A = attn_bf + (size_t)i * 16 * NRES;
  const float* Bsrc = in2d + (size_t)i * NRES * C2;

  // Prologue: stage chunk 0.
#pragma unroll
  for (int t = threadIdx.x; t < (32 * C2) / 4; t += 256)
    lds_copy16(Bsrc + (size_t)t * 4, &buf[0][t * 4]);
  lds_copy_join();

  v8f acc = {};
#pragma unroll 1
  for (int kc = 0; kc < 16; ++kc) {
    int cur = kc & 1;
    if (kc + 1 < 16) {
      const float* gn = Bsrc + (size_t)(kc + 1) * 32 * C2;
#pragma unroll
      for (int t = threadIdx.x; t < (32 * C2) / 4; t += 256)
        lds_copy16(gn + (size_t)t * 4, &buf[cur ^ 1][t * 4]);
    }
    v16bf a = load_a_rm_bf(A + kc * 32, NRES);
    v16bf b = load_b_rm_f32(&buf[cur][c0], C2);
    acc = WMMA_BF16(a, b, acc);
    lds_copy_join();
  }

  int n = lane & 15;
#pragma unroll
  for (int r = 0; r < 8; ++r) {
    int hh = r + half * 8;
    if (hh < NHEAD)
      FA[(size_t)i * NFIN + 576 + hh * C2 + c0 + n] = (bf16)acc[r];
  }
}

// ---------------------------------------------------------------------------
// Output projection: out(512x384) = FA(512x2112) @ wout(2112x384) + bout
// ---------------------------------------------------------------------------
__global__ void k_final(const bf16* __restrict__ FA, const bf16* __restrict__ Wout,
                        const float* __restrict__ bout, float* __restrict__ out) {
  int wid = (blockIdx.x * blockDim.x + threadIdx.x) >> 5;  // 768 waves
  int mt = wid / 24, nt = wid % 24;
  if (mt >= 32) return;
  int lane = threadIdx.x & 31, half = lane >> 4;
  v8f acc = {};
#pragma unroll 1
  for (int kc = 0; kc < 66; ++kc) {
    v16bf a = load_a_rm_bf(FA + (size_t)(mt * 16) * NFIN + kc * 32, NFIN);
    v16bf b = load_b_rm_bf(Wout + (size_t)(kc * 32) * NOUT + nt * 16, NOUT);
    acc = WMMA_BF16(a, b, acc);
  }
  int n = nt * 16 + (lane & 15);
  float bb = bout[n];
#pragma unroll
  for (int r = 0; r < 8; ++r) {
    int m = mt * 16 + r + half * 8;
    out[(size_t)m * NOUT + n] = acc[r] + bb;
  }
}

// ---------------------------------------------------------------------------
// Host launch
// ---------------------------------------------------------------------------
extern "C" void kernel_launch(void* const* d_in, const int* in_sizes, int n_in,
                              void* d_out, int out_size, void* d_ws, size_t ws_size,
                              hipStream_t stream) {
  const float* inputs_1d = (const float*)d_in[0];
  const float* inputs_2d = (const float*)d_in[1];
  const float* mask      = (const float*)d_in[2];
  const float* rot       = (const float*)d_in[3];
  const float* trans     = (const float*)d_in[4];
  const float* wq        = (const float*)d_in[5];
  const float* bq        = (const float*)d_in[6];
  const float* wkv       = (const float*)d_in[7];
  const float* bkv       = (const float*)d_in[8];
  const float* wqp       = (const float*)d_in[9];
  const float* bqp       = (const float*)d_in[10];
  const float* wkvp      = (const float*)d_in[11];
  const float* bkvp      = (const float*)d_in[12];
  const float* w2d       = (const float*)d_in[13];
  const float* b2d       = (const float*)d_in[14];
  const float* tpw       = (const float*)d_in[15];
  const float* wout      = (const float*)d_in[16];
  const float* bout      = (const float*)d_in[17];
  float* out = (float*)d_out;

  char* ws = (char*)d_ws;
  size_t off = 0;
  auto take = [&](size_t bytes) -> char* {
    char* p = ws + off;
    off += (bytes + 255) & ~(size_t)255;
    return p;
  };

  bf16*  Xbf     = (bf16*) take((size_t)NRES * C1 * 2);
  bf16*  Wallbf  = (bf16*) take((size_t)C1 * NPROJ * 2);
  bf16*  W2dbf   = (bf16*) take((size_t)C2 * 16 * 2);
  bf16*  Woutbf  = (bf16*) take((size_t)NFIN * NOUT * 2);
  float* Yproj   = (float*)take((size_t)NRES * NPROJ * 4);
  bf16*  QA      = (bf16*) take((size_t)NHEAD * NRES * 32 * 2);
  bf16*  KB      = (bf16*) take((size_t)NHEAD * NRES * 32 * 2);
  float* Gq      = (float*)take((size_t)NHEAD * NRES * 4);
  float* Gk      = (float*)take((size_t)NHEAD * NRES * 4);
  bf16*  Vs      = (bf16*) take((size_t)NHEAD * NRES * 16 * 2);
  bf16*  Vp      = (bf16*) take((size_t)NHEAD * NRES * 32 * 2);
  float* a2d     = (float*)take((size_t)NHEAD * NRES * NRES * 4);
  bf16*  attn_bf = (bf16*) take((size_t)NRES * 16 * NRES * 2);
  float* RPG     = (float*)take((size_t)NRES * NHEAD * 24 * 4);
  bf16*  FA      = (bf16*) take((size_t)NRES * NFIN * 2);

  const int B = 256;
  // prep / conversions
  k_f2bf<<<(NRES * C1 + B - 1) / B, B, 0, stream>>>(inputs_1d, Xbf, NRES * C1);
  k_f2bf<<<(NFIN * NOUT + B - 1) / B, B, 0, stream>>>(wout, Woutbf, NFIN * NOUT);
  k_build_wall<<<(C1 * NPROJ + B - 1) / B, B, 0, stream>>>(wq, wkv, wqp, wkvp, Wallbf);
  k_build_w2d<<<(C2 * 16 + B - 1) / B, B, 0, stream>>>(w2d, W2dbf);
  k_zero_attn_pad<<<(NRES * 4 * NRES + B - 1) / B, B, 0, stream>>>(attn_bf);

  // projections
  k_proj<<<288, B, 0, stream>>>(Xbf, Wallbf, bq, bkv, bqp, bkvp, Yproj);
  k_post_proj<<<(NHEAD * NRES + B - 1) / B, B, 0, stream>>>(
      Yproj, rot, trans, tpw, QA, KB, Gq, Gk, Vs, Vp);

  // pair bias (first inputs_2d stream, async LDS staging)
  k_a2d<<<2048, B, 0, stream>>>(inputs_2d, W2dbf, b2d, a2d);

  // attention + fused attn@V
  k_attn<<<NHEAD * 32, B, 0, stream>>>(QA, KB, Gq, Gk, a2d, mask, tpw, Vs, Vp,
                                       attn_bf, FA, RPG);

  // local-frame points / norms, and attn@inputs_2d (second inputs_2d stream,
  // double-buffered async LDS staging)
  k_rpl<<<(NRES * 96 + B - 1) / B, B, 0, stream>>>(RPG, rot, trans, FA);
  k_over2d<<<NRES, B, 0, stream>>>(attn_bf, inputs_2d, FA);

  // output projection
  k_final<<<96, B, 0, stream>>>(FA, Woutbf, bout, out);

  (void)in_sizes; (void)n_in; (void)out_size; (void)ws_size;
}
